// DPRNN_5394478924354
// MI455X (gfx1250) — matrix operand order repaired and checked by reading, not compiled, not run on hardware
//
#include <hip/hip_runtime.h>

typedef __attribute__((ext_vector_type(16))) _Float16 v16h;
typedef __attribute__((ext_vector_type(8)))  float    v8f;
typedef __attribute__((ext_vector_type(2)))  float    f32x2;   // clang vector (nontemporal-ok)

#define B_DIM 65536
#define T_DIM 50
#define I_DIM 2
#define H_DIM 20
#define O_DIM 2

union BPack { unsigned int u[8]; v16h v; };

static __device__ __forceinline__ unsigned pkh(float a, float b) {
  unsigned short ua = __builtin_bit_cast(unsigned short, (_Float16)a);
  unsigned short ub = __builtin_bit_cast(unsigned short, (_Float16)b);
  return (unsigned)ua | ((unsigned)ub << 16);
}

// Hardware tanh: single TRANS32 op on gfx1250 (no libm expansion, no divergence).
static __device__ __forceinline__ float htanh(float x) {
#if __has_builtin(__builtin_amdgcn_tanhf)
  return __builtin_amdgcn_tanhf(x);
#elif __has_builtin(__builtin_amdgcn_tanh_f32)
  return __builtin_amdgcn_tanh_f32(x);
#else
  float y;
  // v_nop satisfies the TRANS result-use hazard (1 op required after a TRANS).
  asm volatile("v_tanh_f32 %0, %1\n\tv_nop" : "=v"(y) : "v"(x));
  return y;
#endif
}

// Augmented weight matrix W_aug[m][k]: k<20 -> W_hh[m][k], k=20/21 -> W_ih[m][:],
// k=22 -> b_ih[m]+b_hh[m] (multiplied by the constant-1 channel), else 0.
static __device__ __forceinline__ float waug(const float* Whh, const float* Wih,
                                             const float* bih, const float* bhh,
                                             int m, int k) {
  if (m >= H_DIM) return 0.0f;
  if (k < H_DIM)      return Whh[m * H_DIM + k];
  if (k == H_DIM)     return Wih[m * I_DIM + 0];
  if (k == H_DIM + 1) return Wih[m * I_DIM + 1];
  if (k == H_DIM + 2) return bih[m] + bhh[m];
  return 0.0f;
}

__global__ __launch_bounds__(256) void rnn_wmma_kernel(
    const float* __restrict__ x,    const float* __restrict__ Wih,
    const float* __restrict__ Whh,  const float* __restrict__ bih,
    const float* __restrict__ bhh,  const float* __restrict__ Wout,
    const float* __restrict__ bout, const float* __restrict__ mask,
    float* __restrict__ out)
{
  const int lane = threadIdx.x & 31;   // wave32
  const int wave = threadIdx.x >> 5;
  const int g    = lane >> 4;          // half-wave group
  const int r    = lane & 15;          // batch row within tile / K-pair column
  const long long b0 = ((long long)blockIdx.x * (blockDim.x >> 5) + wave) * 16;
  if (b0 >= B_DIM) return;

  // ---- Constant A matrices (weights), built once per wave ----
  // A order used here (16x32 f16, k = 16g + s, same as B): lane(g,r) holds row r; g=0 -> K{0..15}, g=1 -> K{16..31}
  BPack A0, A1;
#pragma unroll
  for (int j = 0; j < 8; ++j) {
    int kb = g * 16 + 2 * j;
    A0.u[j] = pkh(waug(Whh, Wih, bih, bhh, r,      kb),
                  waug(Whh, Wih, bih, bhh, r,      kb + 1));
    A1.u[j] = pkh(waug(Whh, Wih, bih, bhh, 16 + r, kb),
                  waug(Whh, Wih, bih, bhh, 16 + r, kb + 1));
  }

  // Per-lane output-projection weights: lane(g,r) holds hidden 8g..8g+7 (+ tail 16..19)
  float w0[8], w1[8], w0b[4], w1b[4];
#pragma unroll
  for (int i = 0; i < 8; ++i) { w0[i] = Wout[8 * g + i]; w1[i] = Wout[H_DIM + 8 * g + i]; }
#pragma unroll
  for (int i = 0; i < 4; ++i) { w0b[i] = Wout[16 + i];   w1b[i] = Wout[H_DIM + 16 + i]; }
  const float bo0 = bout[0], bo1 = bout[1];

  const float* xrow = x    + ((long long)(b0 + r) * T_DIM) * I_DIM;
  const float* mrow = mask + ((long long)(b0 + r) * T_DIM) * H_DIM;
  float*       orow = out  + ((long long)(b0 + r) * T_DIM) * O_DIM;

  // h state carried as packed f16 pairs in B layout; h0 = 0
  unsigned p0 = 0, p1 = 0, p2 = 0, p3 = 0;           // own hidden 8g..8g+7 pairs
  unsigned sp0 = 0, sp1 = 0, sp2 = 0, sp3 = 0;       // partner's pairs (xor16)
  unsigned sq0 = 0, sq1 = 0;                         // hidden 16..19 pairs (xor16)
  const unsigned ONEPK = 0x00003C00u;                // (f16)1.0 low, 0 high
  const v8f cz = {};

  // Software pipeline: preload step-0 operands.
  float2 xt  = *(const float2*)(xrow);
  float4 m0v = *(const float4*)(mrow + 8 * g);
  float4 m1v = *(const float4*)(mrow + 8 * g + 4);
  float4 m2v = *(const float4*)(mrow + 16);

  for (int t = 0; t < T_DIM; ++t) {
    // Issue next step's streaming loads early (overlap with WMMA/tanh below).
    const int tn = (t + 1 < T_DIM) ? (t + 1) : t;
    float2 xt_n  = *(const float2*)(xrow + tn * I_DIM);
    float4 m0n   = *(const float4*)(mrow + tn * H_DIM + 8 * g);
    float4 m1n   = *(const float4*)(mrow + tn * H_DIM + 8 * g + 4);
    float4 m2n   = *(const float4*)(mrow + tn * H_DIM + 16);

    unsigned xpk = pkh(xt.x, xt.y);

    // B (32x16 f16): lane(0,r) K0..15 = h0..15; lane(1,r) K16..31 = h16..19,x0,x1,1,0...
    BPack Bm;
    Bm.u[0] = g ? sq0   : p0;
    Bm.u[1] = g ? sq1   : p1;
    Bm.u[2] = g ? xpk   : p2;
    Bm.u[3] = g ? ONEPK : p3;
    Bm.u[4] = g ? 0u : sp0;
    Bm.u[5] = g ? 0u : sp1;
    Bm.u[6] = g ? 0u : sp2;
    Bm.u[7] = g ? 0u : sp3;

    // D = W_aug x state^T : fused  xp + h @ W_hh^T  (C = 0, biases in K=22 channel)
    v8f d0 = __builtin_amdgcn_wmma_f32_16x16x32_f16(false, A0.v, false, Bm.v,
                                                    (short)0, cz, false, false);
    v8f d1 = __builtin_amdgcn_wmma_f32_16x16x32_f16(false, A1.v, false, Bm.v,
                                                    (short)0, cz, false, false);

    // Hardware tanh -> new hidden state. lane(g,r): d0[v] = hidden v+8g of batch r;
    // d1[v] (g=0, v<4) = hidden 16+v of batch r; all other tail rows are exactly 0.
    float th[8];
#pragma unroll
    for (int i = 0; i < 8; ++i) th[i] = htanh(d0[i]);
    float th2[4];
#pragma unroll
    for (int i = 0; i < 4; ++i) th2[i] = htanh(d1[i]);

    // repack for next step's B; half-wave swap via xor-16 shuffles (no LDS transpose)
    p0 = pkh(th[0], th[1]); p1 = pkh(th[2], th[3]);
    p2 = pkh(th[4], th[5]); p3 = pkh(th[6], th[7]);
    unsigned q0 = pkh(th2[0], th2[1]), q1 = pkh(th2[2], th2[3]);
    sp0 = __shfl_xor(p0, 16, 32); sp1 = __shfl_xor(p1, 16, 32);
    sp2 = __shfl_xor(p2, 16, 32); sp3 = __shfl_xor(p3, 16, 32);
    sq0 = __shfl_xor(q0, 16, 32); sq1 = __shfl_xor(q1, 16, 32);

    // ---- dropout + output projection (streams the 262MB mask: the HBM bound) ----
    float mk[8] = {m0v.x, m0v.y, m0v.z, m0v.w, m1v.x, m1v.y, m1v.z, m1v.w};
    float acc0 = 0.f, acc1 = 0.f;
#pragma unroll
    for (int i = 0; i < 8; ++i) {
      float hm = th[i] * mk[i];
      acc0 = fmaf(hm, w0[i], acc0);
      acc1 = fmaf(hm, w1[i], acc1);
    }
    float mb[4] = {m2v.x, m2v.y, m2v.z, m2v.w};
#pragma unroll
    for (int i = 0; i < 4; ++i) {          // th2 == 0 on g=1 lanes, so safe unconditionally
      float hm = th2[i] * mb[i];
      acc0 = fmaf(hm, w0b[i], acc0);
      acc1 = fmaf(hm, w1b[i], acc1);
    }
    acc0 += __shfl_xor(acc0, 16, 32);      // combine hidden halves
    acc1 += __shfl_xor(acc1, 16, 32);
    if (g == 0) {
      f32x2 o; o.x = acc0 + bo0; o.y = acc1 + bo1;
      __builtin_nontemporal_store(o, (f32x2*)(orow + t * O_DIM));  // write-once stream
    }

    // rotate prefetched operands into place
    xt = xt_n; m0v = m0n; m1v = m1n; m2v = m2n;
  }
}

extern "C" void kernel_launch(void* const* d_in, const int* in_sizes, int n_in,
                              void* d_out, int out_size, void* d_ws, size_t ws_size,
                              hipStream_t stream) {
  const float* x    = (const float*)d_in[0];
  const float* Wih  = (const float*)d_in[1];
  const float* Whh  = (const float*)d_in[2];
  const float* bih  = (const float*)d_in[3];
  const float* bhh  = (const float*)d_in[4];
  const float* Wout = (const float*)d_in[5];
  const float* bout = (const float*)d_in[6];
  const float* mask = (const float*)d_in[7];
  float* out = (float*)d_out;

  const int threads = 256;                       // 8 waves/block (wave32)
  const int tiles   = B_DIM / 16;                // 4096 batch tiles
  const int blocks  = tiles / (threads / 32);    // 512
  rnn_wmma_kernel<<<blocks, threads, 0, stream>>>(x, Wih, Whh, bih, bhh,
                                                  Wout, bout, mask, out);
}
